// FactEncoder_81295140979384
// MI455X (gfx1250) — compile-verified, matrix operand order checked by reading, not run on hardware
//
#include <hip/hip_runtime.h>
#include <cstdint>

// ---------------------------------------------------------------------------
// Bidirectional LSTM (FactEncoder) for MI455X / gfx1250, wave32 + WMMA bf16.
//   B=16 F=32 -> N=512 sequences, T=64, D=H=256, 4H=1024, 2 directions.
// Phase 0: swizzle W_f/W_b/U_f/U_b (f32) -> bf16 WMMA B-fragment layout.
// Phase 1: Z[dir] = x(dir-ordered) @ W + b   (big parallel GEMM, WMMA bf16)
// Phase 2: per-16-row persistent blocks run the 64-step recurrence h@U with
//          h kept in LDS in A-fragment layout; gates -> sigmoid/tanh -> out.
// ---------------------------------------------------------------------------

typedef __attribute__((ext_vector_type(16))) __bf16 v16bf;
typedef __attribute__((ext_vector_type(8)))  __bf16 v8bf;
typedef __attribute__((ext_vector_type(8)))  float  v8f;

static constexpr int T_   = 64;
static constexpr int D_   = 256;   // input dim  (= 8 k-tiles of 32)
static constexpr int H_   = 256;   // hidden dim (= 8 k-tiles of 32)
static constexpr int N_   = 512;   // B*F sequences
static constexpr int G4H  = 1024;  // 4*H gate width (= 64 n-tiles of 16)
static constexpr int NT_  = 64;    // n-tiles in 4H
static constexpr int KT_  = 8;     // k-tiles in D or H
// swizzled matrix: [kt][nt][lane(32)][elem(16)] bf16
static constexpr int MAT_ELEMS = KT_ * NT_ * 32 * 16;        // 262144
// swizzled Z per dir: [rowtile(2048)][coltile(64)][lane(32)][8] bf16
static constexpr size_t ZELEMS = (size_t)(T_ * N_ / 16) * NT_ * 32 * 8; // 33.5M

__device__ __forceinline__ __bf16 f2bf(float f) {
  unsigned u = __builtin_bit_cast(unsigned, f);
  unsigned r = u + 0x7FFFu + ((u >> 16) & 1u);          // round-nearest-even
  return __builtin_bit_cast(__bf16, (unsigned short)(r >> 16));
}
__device__ __forceinline__ float bf2f(__bf16 b) {
  unsigned short h = __builtin_bit_cast(unsigned short, b);
  return __builtin_bit_cast(float, (unsigned)h << 16);
}
__device__ __forceinline__ float sigmoidf_(float x) {
  return 1.0f / (1.0f + __expf(-x));
}

// ISA 16-bit fragment position: element (k' in 0..31, mn in 0..15) lives in
// lane = hi*16 + mn, half-word e of the lane's 16 values.
//   vgpr 0..3 : k' = 8*hi + 2*j + b ; vgpr 4..7 : k' = 16 + 8*hi + 2*(j-4) + b
__device__ __forceinline__ void frag_pos(int kk, int mn, int& lane, int& e) {
  int hi = (kk >> 3) & 1;
  int v  = ((kk >> 4) << 2) | ((kk >> 1) & 3);
  e    = (v << 1) | (kk & 1);
  lane = (hi << 4) | mn;
}

// ---------------------------------------------------------------------------
// Kernel 0: f32 [256][1024] -> bf16 fragment-swizzled (B-operand layout).
// mats: 0=W_f 1=W_b 2=U_f 3=U_b
// ---------------------------------------------------------------------------
__global__ __launch_bounds__(256)
void convert_weights(const float* __restrict__ Wf, const float* __restrict__ Wb,
                     const float* __restrict__ Uf, const float* __restrict__ Ub,
                     __bf16* __restrict__ Wsw, __bf16* __restrict__ Usw) {
  int idx = blockIdx.x * 256 + threadIdx.x;      // 4 * 262144 total
  int mat = idx >> 18;
  int el  = idx & ((1 << 18) - 1);
  int k = el >> 10, n = el & 1023;
  const float* src = (mat == 0) ? Wf : (mat == 1) ? Wb : (mat == 2) ? Uf : Ub;
  __bf16* dst = ((mat < 2) ? Wsw : Usw) + (size_t)(mat & 1) * MAT_ELEMS;
  int lane, e;
  frag_pos(k & 31, n & 15, lane, e);
  dst[((size_t)(((k >> 5) * NT_) + (n >> 4)) * 32 + lane) * 16 + e] =
      f2bf(src[(size_t)k * G4H + n]);
}

// ---------------------------------------------------------------------------
// Kernel 1: Z[dir][t][n][4H] = x[t_src][n][:] @ W[dir] + b[dir]   (bf16 WMMA)
// Block tile: 128 rows x 64 cols; 8 waves, each 16 rows x 4 n-tiles.
// dir 0 reads time-reversed x (reference's "forward" branch).
// ---------------------------------------------------------------------------
__global__ __launch_bounds__(256)
void input_proj(const float* __restrict__ facts,
                const __bf16* __restrict__ Wsw,
                const float* __restrict__ b_f, const float* __restrict__ b_b,
                __bf16* __restrict__ Zsw) {
  __shared__ __bf16 As[8 * 32 * 16];   // 8 row-tiles in A-fragment layout, 8 KB
  const int dir  = blockIdx.z;
  const int row0 = blockIdx.x * 128;   // Z row = s*512 + n  (s constant/block)
  const int n0   = blockIdx.y * 64;
  const int tid  = threadIdx.x;
  const int wave = tid >> 5, lane = tid & 31;
  const __bf16* W    = Wsw + (size_t)dir * MAT_ELEMS;
  const float*  bias = dir ? b_b : b_f;

  // Each thread stages 16 contiguous f32 of one row per k-step.
  const int arow = tid >> 1;                 // 0..127 local row
  const int kw   = (tid & 1) * 16;           // low/high half of k-tile
  const int grow = row0 + arow;
  const int nseq = grow & (N_ - 1);
  const int s    = grow >> 9;                // /512
  const int tsrc = (dir == 0) ? (T_ - 1 - s) : s;
  const float* asrc = facts + ((size_t)nseq * T_ + tsrc) * D_;

  v8f acc[4] = {};
  for (int kt = 0; kt < KT_; ++kt) {
    __syncthreads();
    const float* p = asrc + kt * 32 + kw;
#pragma unroll
    for (int i = 0; i < 16; ++i) {
      int lane2, e;
      frag_pos(kw + i, arow & 15, lane2, e);
      As[((arow >> 4) * 32 + lane2) * 16 + e] = f2bf(p[i]);
    }
    __syncthreads();
    const v16bf a = *reinterpret_cast<const v16bf*>(&As[(wave * 32 + lane) * 16]);
#pragma unroll
    for (int j = 0; j < 4; ++j) {
      const v16bf b = *reinterpret_cast<const v16bf*>(
          W + ((size_t)(kt * NT_ + (n0 >> 4) + j) * 32 + lane) * 16);
      acc[j] = __builtin_amdgcn_wmma_f32_16x16x32_bf16(
          false, a, false, b, (short)0, acc[j], false, false);
    }
  }

  // bias + store Z in C-fragment layout (lane -> n, vgpr -> m / m+8)
  const int rowtile = (row0 >> 4) + wave;
#pragma unroll
  for (int j = 0; j < 4; ++j) {
    const float bv = bias[n0 + j * 16 + (lane & 15)];
    v8bf o;
#pragma unroll
    for (int v = 0; v < 8; ++v) o[v] = f2bf(acc[j][v] + bv);
    __bf16* zp = Zsw + (size_t)dir * ZELEMS +
                 ((size_t)((size_t)rowtile * NT_ + (n0 >> 4) + j) * 32 + lane) * 8;
    *reinterpret_cast<v8bf*>(zp) = o;
  }
}

// ---------------------------------------------------------------------------
// Kernel 2: recurrence. One block owns 16 sequence rows for all 64 steps.
// Per step: G = h@U + Z[s]  (512 WMMA / block), then gate math in LDS.
// LDS: hA 8 KB (A-frag bf16) + c 16 KB (f32) + G 32 KB (bf16) = 56 KB.
// ---------------------------------------------------------------------------
__global__ __launch_bounds__(256)
void lstm_recurrent(const __bf16* __restrict__ Zsw,
                    const __bf16* __restrict__ Usw,
                    const unsigned char* __restrict__ mask,
                    float* __restrict__ out) {
  __shared__ __bf16 hA[KT_ * 32 * 16];
  __shared__ float  cSt[16 * H_];
  __shared__ __bf16 Gs[16 * G4H];
  const int dir   = blockIdx.y;
  const int nrow0 = blockIdx.x * 16;
  const int tid = threadIdx.x, wave = tid >> 5, lane = tid & 31;
  const __bf16* U = Usw + (size_t)dir * MAT_ELEMS;
  const __bf16* Z = Zsw + (size_t)dir * ZELEMS;

  for (int i = tid; i < KT_ * 32 * 16; i += 256) hA[i] = f2bf(0.0f);
  for (int i = tid; i < 16 * H_;       i += 256) cSt[i] = 0.0f;
  __syncthreads();

  const int em  = tid >> 4;          // elementwise: row m 0..15
  const int ej0 = (tid & 15) * 16;   // 16 consecutive hidden cols

  for (int s = 0; s < T_; ++s) {
    const size_t rowtile = (size_t)s * (N_ / 16) + blockIdx.x;
    v8f acc[8];
#pragma unroll
    for (int j = 0; j < 8; ++j) {    // init accumulators from Z fragments
      const v8bf z = *reinterpret_cast<const v8bf*>(
          Z + ((rowtile * NT_ + wave * 8 + j) * 32 + lane) * 8);
#pragma unroll
      for (int v = 0; v < 8; ++v) acc[j][v] = bf2f(z[v]);
    }
    if (s + 1 < T_)                  // pull next step's Z toward the caches
      __builtin_prefetch(Z + (((rowtile + N_ / 16) * NT_ + wave * 8) * 32 + lane) * 8, 0, 0);

#pragma unroll
    for (int kt = 0; kt < KT_; ++kt) {
      const v16bf a = *reinterpret_cast<const v16bf*>(&hA[(kt * 32 + lane) * 16]);
#pragma unroll
      for (int j = 0; j < 8; ++j) {
        const v16bf b = *reinterpret_cast<const v16bf*>(
            U + ((size_t)(kt * NT_ + wave * 8 + j) * 32 + lane) * 16);
        acc[j] = __builtin_amdgcn_wmma_f32_16x16x32_bf16(
            false, a, false, b, (short)0, acc[j], false, false);
      }
    }

    // scatter gate pre-activations to LDS (C-frag -> row major)
#pragma unroll
    for (int j = 0; j < 8; ++j) {
      const int col   = (wave * 8 + j) * 16 + (lane & 15);
      const int mbase = (lane >> 4) * 8;
#pragma unroll
      for (int v = 0; v < 8; ++v) Gs[(mbase + v) * G4H + col] = f2bf(acc[j][v]);
    }
    __syncthreads();

    // gate math, masking, state update, output
    const int  tsrc  = (dir == 0) ? (T_ - 1 - s) : s;
    const int  tout  = (dir == 0) ? s : (T_ - 1 - s);
    const int  nglob = nrow0 + em;
    const bool mk    = mask[(size_t)nglob * T_ + tsrc] != 0;
    float* op = out + ((size_t)nglob * T_ + tout) * (2 * H_) + dir * H_;
#pragma unroll
    for (int jj = 0; jj < 16; ++jj) {
      const int j = ej0 + jj;
      const float gi = bf2f(Gs[em * G4H + j]);
      const float gf = bf2f(Gs[em * G4H + H_ + j]);
      const float gg = bf2f(Gs[em * G4H + 2 * H_ + j]);
      const float go = bf2f(Gs[em * G4H + 3 * H_ + j]);
      const float cold = cSt[em * H_ + j];
      int lane2, e;
      frag_pos(j & 31, em, lane2, e);
      const int hidx = ((j >> 5) * 32 + lane2) * 16 + e;
      const float hold = bf2f(hA[hidx]);
      float cn = sigmoidf_(gf) * cold + sigmoidf_(gi) * tanhf(gg);
      float hn = sigmoidf_(go) * tanhf(cn);
      if (!mk) { cn = cold; hn = hold; }   // Keras masking: carry state
      cSt[em * H_ + j] = cn;
      hA[hidx] = f2bf(hn);
      op[j] = hn;
    }
    __syncthreads();
  }
}

// ---------------------------------------------------------------------------
extern "C" void kernel_launch(void* const* d_in, const int* in_sizes, int n_in,
                              void* d_out, int out_size, void* d_ws, size_t ws_size,
                              hipStream_t stream) {
  const float*         facts = (const float*)d_in[0];
  const unsigned char* mask  = (const unsigned char*)d_in[1];  // jnp bool = i8
  const float* W_f = (const float*)d_in[2];
  const float* U_f = (const float*)d_in[3];
  const float* b_f = (const float*)d_in[4];
  const float* W_b = (const float*)d_in[5];
  const float* U_b = (const float*)d_in[6];
  const float* b_b = (const float*)d_in[7];
  float* out = (float*)d_out;

  char* ws = (char*)d_ws;
  __bf16* Wsw = (__bf16*)ws;                                      // 1 MiB
  __bf16* Usw = (__bf16*)(ws + (size_t)2 * MAT_ELEMS * 2);        // 1 MiB
  __bf16* Zsw = (__bf16*)(ws + (size_t)4 * MAT_ELEMS * 2);        // 128 MiB

  convert_weights<<<(4 * MAT_ELEMS) / 256, 256, 0, stream>>>(
      W_f, W_b, U_f, U_b, Wsw, Usw);

  dim3 g1((T_ * N_) / 128, G4H / 64, 2);
  input_proj<<<g1, 256, 0, stream>>>(facts, Wsw, b_f, b_b, Zsw);

  dim3 g2(N_ / 16, 2);
  lstm_recurrent<<<g2, 256, 0, stream>>>(Zsw, Usw, mask, out);
}
